// SelfAttention_48146583388545
// MI455X (gfx1250) — matrix operand order repaired and checked by reading, not compiled, not run on hardware
//
#include <hip/hip_runtime.h>
#include <hip/hip_bf16.h>

typedef __bf16 v16bf __attribute__((ext_vector_type(16)));
typedef __bf16 v8bf  __attribute__((ext_vector_type(8)));
typedef float  v8f   __attribute__((ext_vector_type(8)));
typedef unsigned int v4u __attribute__((ext_vector_type(4)));
typedef int v8i __attribute__((ext_vector_type(8)));
typedef int v4i __attribute__((ext_vector_type(4)));

#define BATCH 8
#define CC    256
#define CQK   32
#define NPIX  4096
#define NSTAGE (NPIX / 64)     // 64 keys per TDM stage
#define VROW  80               // LDS row pitch in bf16: [64B][16B pad][64B][16B pad]
#define VSUB  40               // bf16 elems from row start to keys-32..63 segment
#define VBUFB (CC * VROW * 2)  // bytes per V staging buffer (40960)

// Cross-half (lane ^ 16) exchange via v_permlanex16_b32 with identity selects.
__device__ __forceinline__ float xhalf16(float v) {
    int i = __float_as_int(v);
    int r = __builtin_amdgcn_permlanex16(i, i, 0x76543210, 0xFEDCBA98, false, false);
    return __int_as_float(r);
}

// TDM: stage one V block (tile [256 ch][64 keys], bf16) into LDS at lds_off.
// D# per CDNA5 ISA: 2-D tensor dim0=4096 keys (stride 4096),
// dim1=256 ch; pad_enable, pad_interval=16 dwords (code 3), pad_amount=4 dwords
// (code 3) -> pad after every 64B streamed => 160B LDS row pitch, 16B-aligned
// segments, bank-conflict-spread rows.
__device__ __forceinline__ void tdm_load_v(unsigned lds_off, unsigned long long gaddr) {
    v4u g0;
    g0[0] = 1u;                                            // count=1, user mode
    g0[1] = lds_off;                                       // LDS byte address
    g0[2] = (unsigned)(gaddr & 0xFFFFFFFFu);               // global_addr[31:0]
    g0[3] = (unsigned)((gaddr >> 32) & 0x01FFFFFFu) | 0x80000000u; // [56:32] | type=2
    v8i g1;
    g1[0] = (int)(0x00010000u | (1u << 20) | (3u << 22) | (3u << 25)); // dsz=2B, pad
    g1[1] = (int)((NPIX & 0xFFFFu) << 16);                 // tensor_dim0 lo16
    g1[2] = (int)(((NPIX >> 16) & 0xFFFFu) | ((CC & 0xFFFFu) << 16)); // dim0 hi, dim1 lo
    g1[3] = (int)(((CC >> 16) & 0xFFFFu) | (64u << 16));   // dim1 hi, tile_dim0=64
    g1[4] = (int)CC;                                       // tile_dim1=256, tile_dim2=0
    g1[5] = (int)NPIX;                                     // tensor_dim0_stride lo32
    g1[6] = 0;                                             // stride hi, dim1_stride
    g1[7] = 0;
    v4i z4 = {0, 0, 0, 0};
    v8i z8 = {0, 0, 0, 0, 0, 0, 0, 0};
    // 6-arg form (amdgpu-toolchain clang): (g0, g1, g2, g3, g4, cpol)
    __builtin_amdgcn_tensor_load_to_lds(g0, g1, z4, z4, z8, 0);
}

// ---------------------------------------------------------------------------
// Kernel 1: fused 1x1-conv projections -> bf16 q/k/v in WMMA-friendly layouts
// ---------------------------------------------------------------------------
__global__ void __launch_bounds__(256) proj_qkv(
    const float* __restrict__ x,
    const float* __restrict__ Wq, const float* __restrict__ bq,
    const float* __restrict__ Wk, const float* __restrict__ bk,
    const float* __restrict__ Wv, const float* __restrict__ bv,
    __bf16* __restrict__ qg, __bf16* __restrict__ kg, __bf16* __restrict__ vg)
{
    __shared__ float xs[CC][32];
    const int b  = blockIdx.x >> 7;
    const int n0 = (blockIdx.x & 127) << 5;
    const int tid = threadIdx.x;
    const int p  = tid & 31;
    const int cg = tid >> 5;

    #pragma unroll 4
    for (int it = 0; it < 32; ++it) {
        int cch = it * 8 + cg;
        xs[cch][p] = x[((size_t)b * CC + cch) * NPIX + n0 + p];
    }
    __syncthreads();

    for (int j = 0; j < 40; ++j) {
        int o = cg * 40 + j;
        const float* w;
        float bias;
        __bf16* dst;
        if (o < 32) {
            w = Wq + o * CC; bias = bq[o];
            dst = qg + ((size_t)(b * NPIX + n0 + p)) * CQK + o;
        } else if (o < 64) {
            int o2 = o - 32;
            w = Wk + o2 * CC; bias = bk[o2];
            dst = kg + ((size_t)(b * NPIX + n0 + p)) * CQK + o2;
        } else {
            int o2 = o - 64;
            w = Wv + o2 * CC; bias = bv[o2];
            dst = vg + ((size_t)(b * CC + o2)) * NPIX + n0 + p;
        }
        float sum = bias;
        #pragma unroll 8
        for (int cch = 0; cch < CC; ++cch) sum += w[cch] * xs[cch][p];
        *dst = (__bf16)sum;
    }
}

// ---------------------------------------------------------------------------
// Kernel 2: flash attention. Block = 8 waves = 2 tile-sets x 4 channel-slices.
// Each wave: 4 query tiles x 64 channels. V staged 64 keys at a time via TDM
// (double-buffered, one barrier per 48 WMMAs); each V A-operand reused 4x.
// ---------------------------------------------------------------------------
__global__ void __launch_bounds__(256) flash_attn(
    const __bf16* __restrict__ qg, const __bf16* __restrict__ kg,
    const __bf16* __restrict__ vg, const float* __restrict__ x,
    const float* __restrict__ gamma, float* __restrict__ out)
{
    __shared__ __bf16 vs[2][CC * VROW];      // 2 x 40KB V staging (only LDS here)
    const int wave = threadIdx.x >> 5;
    const int lane = threadIdx.x & 31;
    const int ts = wave >> 2;                // tile-set 0/1
    const int cs = wave & 3;                 // channel slice 0..3 (64 ch each)
    const int b  = blockIdx.x >> 5;          // 32 blocks per batch
    const int qt0 = (blockIdx.x & 31) * 8 + ts * 4;  // first of 4 query tiles
    const int h = lane >> 4;
    const int c = lane & 15;
    const bool hiH = (h != 0);

    // Q^T B-operands for the 4 query tiles (lane: Cqk dims 16h..16h+15, query c)
    v16bf qB[4];
    #pragma unroll
    for (int t = 0; t < 4; ++t) {
        const __bf16* qb = qg + ((size_t)(b * NPIX + (qt0 + t) * 16 + c)) * CQK + h * 16;
        qB[t] = *(const v16bf*)qb;
    }

    v8f vzero = {0.f, 0.f, 0.f, 0.f, 0.f, 0.f, 0.f, 0.f};
    v8f acc[4][4];
    #pragma unroll
    for (int t = 0; t < 4; ++t)
        #pragma unroll
        for (int g = 0; g < 4; ++g) acc[t][g] = vzero;
    float m[4], l[4];
    #pragma unroll
    for (int t = 0; t < 4; ++t) { m[t] = -1e30f; l[t] = 0.f; }

    const __bf16* kb = kg + ((size_t)b * NPIX) * CQK;
    const unsigned long long vgaddr =
        (unsigned long long)(uintptr_t)(vg + (size_t)b * CC * NPIX);

    // Prime the pipeline: stage keys 0..63 into buffer 0 (wave 0 drives TDM).
    if (wave == 0) tdm_load_v(0u, vgaddr);

    for (int st = 0; st < NSTAGE; ++st) {
        if (wave == 0) __builtin_amdgcn_s_wait_tensorcnt(0);
        __syncthreads();                               // stage st visible to all
        if (wave == 0 && st + 1 < NSTAGE)
            tdm_load_v((unsigned)(((st + 1) & 1) * VBUFB), vgaddr + (st + 1) * 128ull);
        const __bf16* vbuf = vs[st & 1];

        #pragma unroll
        for (int sub = 0; sub < 2; ++sub) {
            const int kc = st * 2 + sub;               // 32-key compute chunk

            // K tiles as A-operands (shared by all 4 query tiles)
            const int key0 = kc * 32 + c;
            const __bf16* k0p = kb + (size_t)key0 * CQK;
            const __bf16* k1p = kb + (size_t)(key0 + 16) * CQK;
            v8bf k0lo = *(const v8bf*)(k0p + h * 16);
            v8bf k0hi = *(const v8bf*)(k0p + h * 16 + 8);
            v8bf k1lo = *(const v8bf*)(k1p + h * 16);
            v8bf k1hi = *(const v8bf*)(k1p + h * 16 + 8);
            v16bf ka0, ka1;
            #pragma unroll
            for (int i = 0; i < 8; ++i) {
                ka0[i] = k0lo[i]; ka0[8 + i] = k0hi[i];
                ka1[i] = k1lo[i]; ka1[8 + i] = k1hi[i];
            }

            // Per-tile S^T, online softmax, P^T operand build (in registers)
            v16bf pb[4];
            #pragma unroll
            for (int t = 0; t < 4; ++t) {
                v8f st0 = __builtin_amdgcn_wmma_f32_16x16x32_bf16(
                    false, ka0, false, qB[t], (short)0, vzero, false, false);
                v8f st1 = __builtin_amdgcn_wmma_f32_16x16x32_bf16(
                    false, ka1, false, qB[t], (short)0, vzero, false, false);
                float tm = fmaxf(st0[0], st1[0]);
                #pragma unroll
                for (int r = 1; r < 8; ++r) tm = fmaxf(tm, fmaxf(st0[r], st1[r]));
                tm = fmaxf(tm, xhalf16(tm));
                const float mnew = fmaxf(m[t], tm);
                const float corr = __expf(m[t] - mnew);
                m[t] = mnew;

                float e0[8], e1[8], ps = 0.f;
                #pragma unroll
                for (int r = 0; r < 8; ++r) {
                    e0[r] = __expf(st0[r] - mnew);
                    e1[r] = __expf(st1[r] - mnew);
                    ps += e0[r] + e1[r];
                }
                l[t] = l[t] * corr + ps + xhalf16(ps);

                #pragma unroll
                for (int g = 0; g < 4; ++g)
                    #pragma unroll
                    for (int r = 0; r < 8; ++r) acc[t][g][r] *= corr;

                #pragma unroll
                for (int r = 0; r < 8; ++r) {
                    float a  = hiH ? xhalf16(e1[r]) : e0[r];
                    float b2 = hiH ? e1[r] : xhalf16(e0[r]);
                    pb[t][r]     = (__bf16)a;
                    pb[t][8 + r] = (__bf16)b2;
                }
            }

            // O^T accumulate: each V A-operand (16 ch x 32 keys) reused by 4 tiles
            #pragma unroll
            for (int g = 0; g < 4; ++g) {
                const __bf16* vp = vbuf + (size_t)(cs * 64 + g * 16 + c) * VROW
                                        + sub * VSUB;
                v8bf vlo = *(const v8bf*)(vp + h * 16);
                v8bf vhi = *(const v8bf*)(vp + h * 16 + 8);
                v16bf va;
                #pragma unroll
                for (int i = 0; i < 8; ++i) { va[i] = vlo[i]; va[8 + i] = vhi[i]; }
                #pragma unroll
                for (int t = 0; t < 4; ++t)
                    acc[t][g] = __builtin_amdgcn_wmma_f32_16x16x32_bf16(
                        false, va, false, pb[t], (short)0, acc[t][g], false, false);
            }
        }
    }

    // Epilogue: out[b][ch][q] = gamma * acc/l + x  (queries across lanes)
    const float gm = gamma[0];
    #pragma unroll
    for (int t = 0; t < 4; ++t) {
        const float linv = 1.f / l[t];
        #pragma unroll
        for (int g = 0; g < 4; ++g) {
            #pragma unroll
            for (int r = 0; r < 8; ++r) {
                const int ch = cs * 64 + g * 16 + r + 8 * h;
                const size_t idx = ((size_t)(b * CC + ch)) * NPIX + (qt0 + t) * 16 + c;
                out[idx] = gm * (acc[t][g][r] * linv) + x[idx];
            }
        }
    }
}

// ---------------------------------------------------------------------------
extern "C" void kernel_launch(void* const* d_in, const int* in_sizes, int n_in,
                              void* d_out, int out_size, void* d_ws, size_t ws_size,
                              hipStream_t stream) {
    const float* x     = (const float*)d_in[0];
    const float* Wq    = (const float*)d_in[1];
    const float* bq    = (const float*)d_in[2];
    const float* Wk    = (const float*)d_in[3];
    const float* bk    = (const float*)d_in[4];
    const float* Wv    = (const float*)d_in[5];
    const float* bv    = (const float*)d_in[6];
    const float* gamma = (const float*)d_in[7];
    float* out = (float*)d_out;

    __bf16* qg = (__bf16*)d_ws;                       // 2 MB
    __bf16* kg = qg + (size_t)BATCH * NPIX * CQK;     // 2 MB
    __bf16* vg = kg + (size_t)BATCH * NPIX * CQK;     // 16 MB

    proj_qkv<<<BATCH * (NPIX / 32), 256, 0, stream>>>(x, Wq, bq, Wk, bk, Wv, bv,
                                                      qg, kg, vg);
    flash_attn<<<(BATCH * (NPIX / 16)) / 8, 256, 0, stream>>>(qg, kg, vg, x, gamma, out);
}
